// CausalSelfAttention_16930761081090
// MI455X (gfx1250) — matrix operand order repaired and checked by reading, not compiled, not run on hardware
//
#include <hip/hip_runtime.h>
#include <hip/hip_bf16.h>
#include <stdint.h>

// ---------------------------------------------------------------------------
// CausalSelfAttention forward for gfx1250 (MI455X), bf16 WMMA / f32 accum.
// B=2, T=2048, C=1024, H=16, D=64.
// Round 3: double-buffered pipelines -- async global->LDS (ASYNCcnt) and TDM
// tensor_load_to_lds (TENSORcnt) for tile t+1 overlap the WMMAs on tile t.
// ---------------------------------------------------------------------------

#define USE_ASYNC 1
#define USE_TDM   1

typedef __bf16 bf16;
typedef __attribute__((ext_vector_type(16))) __bf16 bf16x16;
typedef __attribute__((ext_vector_type(8)))  __bf16 bf16x8;
typedef __attribute__((ext_vector_type(8)))  float  f32x8;
typedef __attribute__((ext_vector_type(8)))  unsigned int u32x8;
typedef __attribute__((ext_vector_type(4)))  unsigned int u32x4;

#define T_SEQ 2048
#define C_DIM 1024
#define H_NUM 16
#define D_DIM 64

// LDS byte offset of a generic pointer into __shared__
#define LDS_OFF(p) \
  ((unsigned)(uintptr_t)(__attribute__((address_space(3))) char*)(void*)(p))

__device__ __forceinline__ f32x8 wmma_bf16(const u32x8& a, const u32x8& b, f32x8 c) {
  return __builtin_amdgcn_wmma_f32_16x16x32_bf16(
      false, __builtin_bit_cast(bf16x16, a),
      false, __builtin_bit_cast(bf16x16, b),
      (short)0, c, false, false);
}

__device__ __forceinline__ f32x8 zero8() {
  f32x8 z = {0.f, 0.f, 0.f, 0.f, 0.f, 0.f, 0.f, 0.f};
  return z;
}

#if USE_ASYNC
// Per-lane 16B async copy global -> LDS (tracked by ASYNCcnt, bypasses VGPRs)
__device__ __forceinline__ void async_copy_b128(unsigned lds_off, const void* g) {
  unsigned long long ga = (unsigned long long)(uintptr_t)g;
  asm volatile("global_load_async_to_lds_b128 %0, %1, off"
               :: "v"(lds_off), "v"(ga) : "memory");
}
__device__ __forceinline__ void wait_asynccnt0() {
  asm volatile("s_wait_asynccnt 0x0" ::: "memory");
}
#endif

#if USE_TDM
// 2D TDM load: tile_d1 rows of tile_d0 elements (2B each), global row stride
// row_stride elements, LDS destination padded by pad_amount_dw DWORDs every
// 2^(pad_interval_code+1) DWORDs (D# layout per CDNA5 ISA ch.8).
__device__ __forceinline__ void tdm_load_2d(unsigned lds_off, const void* gaddr,
                                            unsigned tile_d0, unsigned tile_d1,
                                            unsigned row_stride,
                                            unsigned pad_interval_code,
                                            unsigned pad_amount_dw) {
  uint64_t ga = (uint64_t)(uintptr_t)gaddr;
  u32x4 g0;
  g0[0] = 1u;                                        // count=1, user mode
  g0[1] = lds_off;                                   // lds_addr
  g0[2] = (unsigned)(ga & 0xFFFFFFFFu);              // global_addr[31:0]
  g0[3] = (unsigned)((ga >> 32) & 0x1FFFFFFu) | (2u << 30);  // [56:32], type=2

  u32x8 g1;
  const unsigned pad_en = pad_amount_dw ? 1u : 0u;
  g1[0] = (1u << 16)                                 // data_size = 2 bytes
        | (pad_en << 20)
        | (pad_interval_code << 22)
        | (pad_en ? ((pad_amount_dw - 1u) << 25) : 0u);
  g1[1] = (tile_d0 & 0xFFFFu) << 16;                  // tensor_dim0[15:0]
  g1[2] = (tile_d0 >> 16) | ((tile_d1 & 0xFFFFu) << 16);
  g1[3] = (tile_d1 >> 16) | ((tile_d0 & 0xFFFFu) << 16);  // tile_dim0
  g1[4] = (tile_d1 & 0xFFFFu);                        // tile_dim1, tile_dim2=0
  uint64_t s0 = row_stride;
  uint64_t s1 = (uint64_t)row_stride * tile_d1;
  g1[5] = (unsigned)(s0 & 0xFFFFFFFFu);               // dim0_stride[31:0]
  g1[6] = (unsigned)(s0 >> 32) | ((unsigned)(s1 & 0xFFFFu) << 16);
  g1[7] = (unsigned)(s1 >> 16);
  asm volatile("tensor_load_to_lds %0, %1" :: "s"(g0), "s"(g1) : "memory");
}
#endif

// ---------------------------------------------------------------------------
// fp32 -> bf16 streaming convert
// ---------------------------------------------------------------------------
__global__ void cvt_f32_to_bf16(const float* __restrict__ in,
                                bf16* __restrict__ out, int n) {
  int i = blockIdx.x * blockDim.x + threadIdx.x;
  int stride = gridDim.x * blockDim.x;
  for (; i < n; i += stride) out[i] = (bf16)in[i];
}

// ---------------------------------------------------------------------------
// QKV GEMM + bias + RoPE + q-scale.  Block tile 128x128, BK=32, 8 waves,
// double-buffered LDS pipeline.
// ---------------------------------------------------------------------------
#define GEMM_LDA 40  // padded LDS row (bf16): 80B stride, conflict-free

__global__ __launch_bounds__(256)
void qkv_gemm_rope(const bf16* __restrict__ A, const bf16* __restrict__ Bmat,
                   const float* __restrict__ bias, const float* __restrict__ rc,
                   bf16* __restrict__ qout, bf16* __restrict__ kout,
                   bf16* __restrict__ vout) {
  __shared__ bf16 As[2][128][GEMM_LDA];
  __shared__ bf16 Bs[2][128][GEMM_LDA];
  const int N = 3 * C_DIM, K = C_DIM;
  const int tid  = threadIdx.x;
  const int lane = tid & 31;
  const int wave = tid >> 5;
  const int wm = wave & 3, wn = wave >> 2;
  const int l16 = lane & 15, hh = lane >> 4;
  const int bm = blockIdx.y * 128;
  const int bn = blockIdx.x * 128;
  const int arow = tid >> 1, aseg = tid & 1;   // A staging role
  const int bkk  = tid >> 3, bng  = tid & 7;   // B staging role

  f32x8 acc[2][4];
#pragma unroll
  for (int s = 0; s < 2; ++s)
#pragma unroll
    for (int j = 0; j < 4; ++j) acc[s][j] = zero8();

  // ---- prologue: stage tile 0 into buffer 0
  {
    const bf16* asrc = A + (size_t)(bm + arow) * K + aseg * 16;
#if USE_ASYNC
    async_copy_b128(LDS_OFF(&As[0][arow][aseg * 16]),     (const void*)asrc);
    async_copy_b128(LDS_OFF(&As[0][arow][aseg * 16 + 8]), (const void*)(asrc + 8));
#else
    *(bf16x8*)&As[0][arow][aseg * 16]     = *(const bf16x8*)(asrc);
    *(bf16x8*)&As[0][arow][aseg * 16 + 8] = *(const bf16x8*)(asrc + 8);
#endif
    const bf16* bsrc = Bmat + (size_t)bkk * N + bn + bng * 16;
    bf16x8 w0 = *(const bf16x8*)(bsrc);
    bf16x8 w1 = *(const bf16x8*)(bsrc + 8);
#pragma unroll
    for (int i = 0; i < 8; ++i) {
      Bs[0][bng * 16 + i][bkk]     = w0[i];
      Bs[0][bng * 16 + 8 + i][bkk] = w1[i];
    }
#if USE_ASYNC
    wait_asynccnt0();
#endif
  }
  __syncthreads();

  const int NK = K / 32;
  for (int t = 0; t < NK; ++t) {
    const int buf = t & 1;
    const bool have_next = (t + 1 < NK);
    bf16x8 nw0, nw1;
    if (have_next) {
      // issue tile t+1: async A into alternate buffer, B tile into registers
      const int kb1 = (t + 1) * 32;
      const bf16* asrc = A + (size_t)(bm + arow) * K + kb1 + aseg * 16;
#if USE_ASYNC
      async_copy_b128(LDS_OFF(&As[buf ^ 1][arow][aseg * 16]),     (const void*)asrc);
      async_copy_b128(LDS_OFF(&As[buf ^ 1][arow][aseg * 16 + 8]), (const void*)(asrc + 8));
#else
      *(bf16x8*)&As[buf ^ 1][arow][aseg * 16]     = *(const bf16x8*)(asrc);
      *(bf16x8*)&As[buf ^ 1][arow][aseg * 16 + 8] = *(const bf16x8*)(asrc + 8);
#endif
      const bf16* bsrc = Bmat + (size_t)(kb1 + bkk) * N + bn + bng * 16;
      nw0 = *(const bf16x8*)(bsrc);
      nw1 = *(const bf16x8*)(bsrc + 8);
      if (t + 2 < NK) {  // prefetch one step beyond the pipeline
        __builtin_prefetch(bsrc + 32 * N, 0, 0);
        __builtin_prefetch(asrc + 32, 0, 0);
      }
    }

    // ---- compute on tile t (overlaps in-flight async / global loads)
    u32x8 areg[2];
#pragma unroll
    for (int s = 0; s < 2; ++s) {
      const int m = wm * 32 + s * 16 + l16;
#pragma unroll
      for (int vv = 0; vv < 8; ++vv) {
        const int kidx = ((vv >> 2) * 16) + hh * 8 + ((vv & 3) * 2);
        areg[s][vv] = *(const unsigned int*)&As[buf][m][kidx];
      }
    }
#pragma unroll
    for (int j = 0; j < 4; ++j) {
      u32x8 breg;
      const int n = wn * 64 + j * 16 + l16;
#pragma unroll
      for (int vv = 0; vv < 8; ++vv)
        breg[vv] = *(const unsigned int*)&Bs[buf][n][((vv >> 2) * 16) + hh * 8 + ((vv & 3) * 2)];
      acc[0][j] = wmma_bf16(areg[0], breg, acc[0][j]);
      acc[1][j] = wmma_bf16(areg[1], breg, acc[1][j]);
    }
    __syncthreads();  // readers of tile t done

    if (have_next) {
#if USE_ASYNC
      wait_asynccnt0();  // A tile t+1 resident in LDS
#endif
#pragma unroll
      for (int i = 0; i < 8; ++i) {
        Bs[buf ^ 1][bng * 16 + i][bkk]     = nw0[i];
        Bs[buf ^ 1][bng * 16 + 8 + i][bkk] = nw1[i];
      }
    }
    __syncthreads();  // tile t+1 visible to all waves
  }

  // ---- epilogue: bias + RoPE (q,k) + scale (q); write (B,H,T,D) bf16
#pragma unroll
  for (int s = 0; s < 2; ++s) {
#pragma unroll
    for (int j = 0; j < 4; ++j) {
      const int ncol = bn + wn * 64 + j * 16 + l16;
      const int which = ncol >> 10;      // 0=q 1=k 2=v
      const int rem   = ncol & 1023;
      const int head  = rem >> 6;
      const int d     = rem & 63;
      const float bv  = bias[ncol];
#pragma unroll
      for (int r = 0; r < 8; ++r) {
        const int grow = bm + wm * 32 + s * 16 + r + hh * 8;
        const int bidx = grow >> 11;
        const int t    = grow & (T_SEQ - 1);
        float val = acc[s][j][r] + bv;
        const float partner = __shfl_xor(val, 1);
        float outv;
        if (which < 2) {
          const float* rcp = rc + t * D_DIM + (d & ~1);
          const float sn = rcp[0], cs = rcp[1];
          outv = (d & 1) ? (val * cs + partner * sn)
                         : (val * cs - partner * sn);
          if (which == 0) outv *= 0.125f;  // 1/sqrt(64)
        } else {
          outv = val;
        }
        bf16* dst = (which == 0) ? qout : (which == 1) ? kout : vout;
        dst[(((size_t)bidx * H_NUM + head) * T_SEQ + t) * D_DIM + d] = (bf16)outv;
      }
    }
  }
}

// ---------------------------------------------------------------------------
// Flash attention: block = 128 query rows of one (b,h); 8 waves x 16 rows.
// Double-buffered: TDM K-tile DMA + V-tile regs for block t+1 overlap the
// WMMAs on block t.
// ---------------------------------------------------------------------------
__global__ __launch_bounds__(256)
void attn_flash(const bf16* __restrict__ q, const bf16* __restrict__ k,
                const bf16* __restrict__ v, bf16* __restrict__ attn) {
  __shared__ bf16 kt[2][32][72];     // K tile (kpos, d), 144B row stride
  __shared__ bf16 vt[2][64][40];     // V^T tile (d, kpos), 80B row stride
  __shared__ bf16 pb[8][16][40];     // per-wave P bounce

  const int tid  = threadIdx.x;
  const int lane = tid & 31;
  const int wave = tid >> 5;
  const int l16 = lane & 15, hh = lane >> 4;
  const int wgq = blockIdx.x;
  const int bh  = blockIdx.y;
  const size_t base = (size_t)bh * T_SEQ * D_DIM;
  const int qbase = wgq * 128 + wave * 16;
  const int srow = tid >> 3, sc8 = (tid & 7) * 8;  // staging role

  // Q in A-layout registers (two K=32 chunks covering D=64)
  u32x8 qa[2];
#pragma unroll
  for (int s = 0; s < 2; ++s)
#pragma unroll
    for (int vv = 0; vv < 8; ++vv) {
      const int d = s * 32 + ((vv >> 2) * 16) + hh * 8 + ((vv & 3) * 2);
      qa[s][vv] = *(const unsigned int*)(q + base + (size_t)(qbase + l16) * D_DIM + d);
    }

  f32x8 acc[4];
#pragma unroll
  for (int dt = 0; dt < 4; ++dt) acc[dt] = zero8();
  float mrow[8], lrow[8];
#pragma unroll
  for (int r = 0; r < 8; ++r) { mrow[r] = -1e30f; lrow[r] = 0.f; }

  const int nkb = wgq * 4 + 4;

  // ---- prologue: stage key-block 0 into buffer 0
  {
#if USE_TDM
    if (wave == 0)
      tdm_load_2d(LDS_OFF(&kt[0][0][0]), (const void*)(k + base),
                  64, 32, 64, /*pad_interval=*/4, /*pad_dw=*/4);
#else
    *(bf16x8*)&kt[0][srow][sc8] =
        *(const bf16x8*)(k + base + (size_t)srow * D_DIM + sc8);
#endif
    bf16x8 vv8 = *(const bf16x8*)(v + base + (size_t)srow * D_DIM + sc8);
#pragma unroll
    for (int i = 0; i < 8; ++i) vt[0][sc8 + i][srow] = vv8[i];
#if USE_TDM
    if (wave == 0) __builtin_amdgcn_s_wait_tensorcnt(0);
#endif
  }
  __syncthreads();

  for (int t = 0; t < nkb; ++t) {
    const int buf = t & 1;
    const bool have_next = (t + 1 < nkb);
    bf16x8 nv8;
    if (have_next) {
      const size_t goff = base + (size_t)((t + 1) * 32 + srow) * D_DIM + sc8;
#if USE_TDM
      if (wave == 0)
        tdm_load_2d(LDS_OFF(&kt[buf ^ 1][0][0]),
                    (const void*)(k + base + (size_t)(t + 1) * 32 * D_DIM),
                    64, 32, 64, 4, 4);
#else
      *(bf16x8*)&kt[buf ^ 1][srow][sc8] = *(const bf16x8*)(k + goff);
#endif
      nv8 = *(const bf16x8*)(v + goff);  // V block t+1 held in regs
    }

    if (t * 32 <= qbase + 15) {  // wave-uniform causal skip
      f32x8 S[2];
#pragma unroll
      for (int nh = 0; nh < 2; ++nh) {
        u32x8 br[2];
        const int krow = nh * 16 + l16;
#pragma unroll
        for (int s = 0; s < 2; ++s)
#pragma unroll
          for (int vv = 0; vv < 8; ++vv)
            br[s][vv] = *(const unsigned int*)&kt[buf][krow][s * 32 + ((vv >> 2) * 16) + hh * 8 + ((vv & 3) * 2)];
        f32x8 z = zero8();
        z = wmma_bf16(qa[0], br[0], z);
        z = wmma_bf16(qa[1], br[1], z);
#pragma unroll
        for (int r = 0; r < 8; ++r) {
          const int qrow = qbase + r + hh * 8;
          const int kcol = t * 32 + nh * 16 + l16;
          if (kcol > qrow) z[r] = -1e30f;
        }
        S[nh] = z;
      }

#pragma unroll
      for (int r = 0; r < 8; ++r) {
        float bmax = fmaxf(S[0][r], S[1][r]);
#pragma unroll
        for (int off = 8; off > 0; off >>= 1)
          bmax = fmaxf(bmax, __shfl_xor(bmax, off, 16));
        const float mn = fmaxf(mrow[r], bmax);
        const float alpha = __expf(mrow[r] - mn);
        mrow[r] = mn;
        const float p0 = __expf(S[0][r] - mn);
        const float p1 = __expf(S[1][r] - mn);
        float rs = p0 + p1;
#pragma unroll
        for (int off = 8; off > 0; off >>= 1) rs += __shfl_xor(rs, off, 16);
        lrow[r] = lrow[r] * alpha + rs;
#pragma unroll
        for (int dt = 0; dt < 4; ++dt) acc[dt][r] *= alpha;
        const int m = r + hh * 8;
        pb[wave][m][l16]      = (bf16)p0;
        pb[wave][m][16 + l16] = (bf16)p1;
      }
      asm volatile("s_wait_dscnt 0x0" ::: "memory");

      u32x8 pa;
#pragma unroll
      for (int vv = 0; vv < 8; ++vv) {
        const int kk = ((vv >> 2) * 16) + hh * 8 + ((vv & 3) * 2);
        pa[vv] = *(const unsigned int*)&pb[wave][l16][kk];
      }
#pragma unroll
      for (int dt = 0; dt < 4; ++dt) {
        u32x8 bv;
#pragma unroll
        for (int vv = 0; vv < 8; ++vv)
          bv[vv] = *(const unsigned int*)&vt[buf][dt * 16 + l16][((vv >> 2) * 16) + hh * 8 + ((vv & 3) * 2)];
        acc[dt] = wmma_bf16(pa, bv, acc[dt]);
      }
    }
    __syncthreads();  // readers of block t done

    if (have_next) {
#pragma unroll
      for (int i = 0; i < 8; ++i) vt[buf ^ 1][sc8 + i][srow] = nv8[i];
#if USE_TDM
      if (wave == 0) __builtin_amdgcn_s_wait_tensorcnt(0);
#endif
    }
    __syncthreads();  // block t+1 visible to all waves
  }

  const int b = bh >> 4, hd = bh & 15;
#pragma unroll
  for (int r = 0; r < 8; ++r) {
    const float inv = 1.0f / lrow[r];
    const int t = qbase + r + hh * 8;
    const size_t o = ((size_t)(b * T_SEQ + t)) * C_DIM + hd * D_DIM;
#pragma unroll
    for (int dt = 0; dt < 4; ++dt)
      attn[o + dt * 16 + l16] = (bf16)(acc[dt][r] * inv);
  }
}

// ---------------------------------------------------------------------------
// Output projection: attn_bf16 (4096x1024) @ Wout_bf16 (1024x1024) + bout -> f32
// Double-buffered pipeline, same core as qkv_gemm_rope.
// ---------------------------------------------------------------------------
__global__ __launch_bounds__(256)
void out_proj(const bf16* __restrict__ A, const bf16* __restrict__ Bmat,
              const float* __restrict__ bias, float* __restrict__ out) {
  __shared__ bf16 As[2][128][GEMM_LDA];
  __shared__ bf16 Bs[2][128][GEMM_LDA];
  const int N = C_DIM, K = C_DIM;
  const int tid  = threadIdx.x;
  const int lane = tid & 31;
  const int wave = tid >> 5;
  const int wm = wave & 3, wn = wave >> 2;
  const int l16 = lane & 15, hh = lane >> 4;
  const int bm = blockIdx.y * 128;
  const int bn = blockIdx.x * 128;
  const int arow = tid >> 1, aseg = tid & 1;
  const int bkk  = tid >> 3, bng  = tid & 7;

  f32x8 acc[2][4];
#pragma unroll
  for (int s = 0; s < 2; ++s)
#pragma unroll
    for (int j = 0; j < 4; ++j) acc[s][j] = zero8();

  {
    const bf16* asrc = A + (size_t)(bm + arow) * K + aseg * 16;
#if USE_ASYNC
    async_copy_b128(LDS_OFF(&As[0][arow][aseg * 16]),     (const void*)asrc);
    async_copy_b128(LDS_OFF(&As[0][arow][aseg * 16 + 8]), (const void*)(asrc + 8));
#else
    *(bf16x8*)&As[0][arow][aseg * 16]     = *(const bf16x8*)(asrc);
    *(bf16x8*)&As[0][arow][aseg * 16 + 8] = *(const bf16x8*)(asrc + 8);
#endif
    const bf16* bsrc = Bmat + (size_t)bkk * N + bn + bng * 16;
    bf16x8 w0 = *(const bf16x8*)(bsrc);
    bf16x8 w1 = *(const bf16x8*)(bsrc + 8);
#pragma unroll
    for (int i = 0; i < 8; ++i) {
      Bs[0][bng * 16 + i][bkk]     = w0[i];
      Bs[0][bng * 16 + 8 + i][bkk] = w1[i];
    }
#if USE_ASYNC
    wait_asynccnt0();
#endif
  }
  __syncthreads();

  const int NK = K / 32;
  for (int t = 0; t < NK; ++t) {
    const int buf = t & 1;
    const bool have_next = (t + 1 < NK);
    bf16x8 nw0, nw1;
    if (have_next) {
      const int kb1 = (t + 1) * 32;
      const bf16* asrc = A + (size_t)(bm + arow) * K + kb1 + aseg * 16;
#if USE_ASYNC
      async_copy_b128(LDS_OFF(&As[buf ^ 1][arow][aseg * 16]),     (const void*)asrc);
      async_copy_b128(LDS_OFF(&As[buf ^ 1][arow][aseg * 16 + 8]), (const void*)(asrc + 8));
#else
      *(bf16x8*)&As[buf ^ 1][arow][aseg * 16]     = *(const bf16x8*)(asrc);
      *(bf16x8*)&As[buf ^ 1][arow][aseg * 16 + 8] = *(const bf16x8*)(asrc + 8);
#endif
      const bf16* bsrc = Bmat + (size_t)(kb1 + bkk) * N + bn + bng * 16;
      nw0 = *(const bf16x8*)(bsrc);
      nw1 = *(const bf16x8*)(bsrc + 8);
      if (t + 2 < NK) {
        __builtin_prefetch(bsrc + 32 * N, 0, 0);
        __builtin_prefetch(asrc + 32, 0, 0);
      }
    }

    u32x8 areg[2];
#pragma unroll
    for (int s = 0; s < 2; ++s) {
      const int m = wm * 32 + s * 16 + l16;
#pragma unroll
      for (int vv = 0; vv < 8; ++vv) {
        const int kidx = ((vv >> 2) * 16) + hh * 8 + ((vv & 3) * 2);
        areg[s][vv] = *(const unsigned int*)&As[buf][m][kidx];
      }
    }
#pragma unroll
    for (int j = 0; j < 4; ++j) {
      u32x8 breg;
      const int n = wn * 64 + j * 16 + l16;
#pragma unroll
      for (int vv = 0; vv < 8; ++vv)
        breg[vv] = *(const unsigned int*)&Bs[buf][n][((vv >> 2) * 16) + hh * 8 + ((vv & 3) * 2)];
      acc[0][j] = wmma_bf16(areg[0], breg, acc[0][j]);
      acc[1][j] = wmma_bf16(areg[1], breg, acc[1][j]);
    }
    __syncthreads();

    if (have_next) {
#if USE_ASYNC
      wait_asynccnt0();
#endif
#pragma unroll
      for (int i = 0; i < 8; ++i) {
        Bs[buf ^ 1][bng * 16 + i][bkk]     = nw0[i];
        Bs[buf ^ 1][bng * 16 + 8 + i][bkk] = nw1[i];
      }
    }
    __syncthreads();
  }

#pragma unroll
  for (int s = 0; s < 2; ++s) {
#pragma unroll
    for (int j = 0; j < 4; ++j) {
      const int ncol = bn + wn * 64 + j * 16 + l16;
      const float bv = bias[ncol];
#pragma unroll
      for (int r = 0; r < 8; ++r) {
        const int grow = bm + wm * 32 + s * 16 + r + hh * 8;
        out[(size_t)grow * N + ncol] = acc[s][j][r] + bv;
      }
    }
  }
}

// ---------------------------------------------------------------------------
// Host launcher
// ---------------------------------------------------------------------------
extern "C" void kernel_launch(void* const* d_in, const int* in_sizes, int n_in,
                              void* d_out, int out_size, void* d_ws, size_t ws_size,
                              hipStream_t stream) {
  const float* x    = (const float*)d_in[0];
  const float* rc   = (const float*)d_in[1];
  const float* Wqkv = (const float*)d_in[2];
  const float* bqkv = (const float*)d_in[3];
  const float* Wout = (const float*)d_in[4];
  const float* bout = (const float*)d_in[5];
  float* out = (float*)d_out;

  char* ws = (char*)d_ws;
  bf16* xb    = (bf16*)(ws);
  bf16* wqkvb = (bf16*)(ws + ((size_t)8  << 20));
  bf16* woutb = (bf16*)(ws + ((size_t)14 << 20));
  bf16* qb    = (bf16*)(ws + ((size_t)16 << 20));
  bf16* kb    = (bf16*)(ws + ((size_t)24 << 20));
  bf16* vb    = (bf16*)(ws + ((size_t)32 << 20));
  bf16* ab    = (bf16*)(ws + ((size_t)40 << 20));

  cvt_f32_to_bf16<<<2048, 256, 0, stream>>>(x,    xb,    2 * T_SEQ * C_DIM);
  cvt_f32_to_bf16<<<2048, 256, 0, stream>>>(Wqkv, wqkvb, C_DIM * 3 * C_DIM);
  cvt_f32_to_bf16<<<2048, 256, 0, stream>>>(Wout, woutb, C_DIM * C_DIM);

  qkv_gemm_rope<<<dim3(24, 32), 256, 0, stream>>>(xb, wqkvb, bqkv, rc,
                                                  qb, kb, vb);
  attn_flash<<<dim3(T_SEQ / 128, 2 * H_NUM), 256, 0, stream>>>(qb, kb, vb, ab);
  out_proj<<<dim3(8, 32), 256, 0, stream>>>(ab, woutb, bout, out);
}
